// multi_head_attention_88545045775171
// MI455X (gfx1250) — matrix operand order repaired and checked by reading, not compiled, not run on hardware
//
#include <hip/hip_runtime.h>
#include <hip/hip_bf16.h>
#include <stdint.h>

#define BB 2
#define SS 2048
#define DD 1024
#define HH 16
#define DH 64

typedef __bf16 bf16;
typedef bf16  v16bf __attribute__((ext_vector_type(16)));
typedef bf16  v4bf  __attribute__((ext_vector_type(4)));
typedef bf16  v2bf  __attribute__((ext_vector_type(2)));
typedef float v8f   __attribute__((ext_vector_type(8)));
typedef float v4f_  __attribute__((ext_vector_type(4)));
typedef unsigned int u32x4 __attribute__((ext_vector_type(4)));
typedef int i32x4 __attribute__((ext_vector_type(4)));
typedef int i32x8 __attribute__((ext_vector_type(8)));

__device__ __forceinline__ bf16 tobf(float x) { return (bf16)x; }

__device__ __forceinline__ v8f vzero() {
    v8f z = {0.f, 0.f, 0.f, 0.f, 0.f, 0.f, 0.f, 0.f};
    return z;
}

union FragU { v16bf v; uint4 q[2]; };

// A-fragment 16x32 bf16: lane L holds row (L&15); lanes 0-15 carry K=0..7 &
// 16..23, lanes 16-31 carry K=8..15 & 24..31 (CDNA5 ISA 7.12.2).
__device__ __forceinline__ v16bf load_a(const bf16* base, int ld) {
    int lane = threadIdx.x & 31;
    const bf16* p = base + (lane & 15) * ld + ((lane >> 4) << 3);
    FragU u;
    u.q[0] = *(const uint4*)p;
    u.q[1] = *(const uint4*)(p + 16);
    return u.v;
}

// B-fragment 32x16 bf16 from an [N=16 rows][K] row-major buffer:
// lane L holds column N=(L&15); lanes 0-15 K=0..7 & 16..23, lanes 16-31 K=8..15 & 24..31.
__device__ __forceinline__ v16bf load_b(const bf16* base, int ld) {
    int lane = threadIdx.x & 31;
    const bf16* p = base + (lane & 15) * ld + ((lane >> 4) << 3);
    FragU u;
    u.q[0] = *(const uint4*)p;
    u.q[1] = *(const uint4*)(p + 16);
    return u.v;
}

__device__ __forceinline__ v8f wmma_bf16(v16bf a, v16bf b, v8f c) {
    return __builtin_amdgcn_wmma_f32_16x16x32_bf16(
        false, a, false, b, (short)0, c, false, false);
}

// ---------------------------------------------------------------------------
// Tensor Data Mover: 2D bf16 tile global -> LDS with per-row pad of
// 4 DWORDs every 32 DWORDs (64-elem rows land at stride 72 elems in LDS).
// D# layout per CDNA5 ISA 8.3/8.4.
// ---------------------------------------------------------------------------
__device__ __forceinline__ void tdm_load_2d_bf16(
    uint32_t lds_off, const void* gsrc, uint32_t tile_d0, uint32_t tile_d1,
    uint32_t tdim0, uint32_t tdim1, uint64_t stride0) {
    uint64_t ga = (uint64_t)(uintptr_t)gsrc;
    u32x4 g0;
    g0[0] = 1u;                                   // count=1, user mode
    g0[1] = lds_off;                              // lds_addr
    g0[2] = (uint32_t)ga;                         // global_addr[31:0]
    g0[3] = ((uint32_t)(ga >> 32) & 0x1FFFFFFu) | (2u << 30);  // type=2
    i32x8 g1;
    g1[0] = (int)((1u << 16)      // data_size = 2 bytes (bf16)
                  | (1u << 20)    // pad_enable
                  | (4u << 22)    // pad_interval: 32 DWORDs (one 64-bf16 row)
                  | (3u << 25));  // pad_amount: 4 DWORDs (16B = 8 bf16)
    g1[1] = (int)((tdim0 & 0xFFFFu) << 16);                      // tensor_dim0 lo
    g1[2] = (int)(((tdim0 >> 16) & 0xFFFFu) | ((tdim1 & 0xFFFFu) << 16));
    g1[3] = (int)(((tdim1 >> 16) & 0xFFFFu) | ((tile_d0 & 0xFFFFu) << 16));
    g1[4] = (int)(tile_d1 & 0xFFFFu);                            // tile_dim1, tile_dim2=0
    g1[5] = (int)(uint32_t)stride0;                              // dim0 stride lo
    g1[6] = (int)((uint32_t)(stride0 >> 32) & 0xFFFFu);          // dim0 stride hi
    g1[7] = 0;
    i32x4 z4 = {0, 0, 0, 0};
#if defined(__clang_major__) && (__clang_major__ >= 23)
    i32x8 z8 = {0, 0, 0, 0, 0, 0, 0, 0};
    __builtin_amdgcn_tensor_load_to_lds(g0, g1, z4, z4, z8, 0);
#else
    __builtin_amdgcn_tensor_load_to_lds(g0, g1, z4, z4, 0);
#endif
}

// ---------------------------------------------------------------------------
// Generic 128x128-tile GEMM, K-chunk 64, register-staged prefetch pipeline:
//   stage top: convert+store staged regs -> LDS; then issue next chunk's
//   global loads; they complete while the 16 WMMAs of this chunk run.
// MODE 0: q   -> bf16 [H,B,S,64] + f32 [H,B,S,64]
// MODE 1: k   -> bf16 [H,B,S,64]
// MODE 2: v   -> bf16 [H,B,64,S]   (transposed for P@V B-fragments)
// MODE 3: res -> f32  [B*S,1024] = X + tanh(Y)
// ---------------------------------------------------------------------------
#define GK  64
#define GLD 72

template <int MODE>
__global__ __launch_bounds__(256) void gemm128(const float* __restrict__ X,
                                               const float* __restrict__ W,
                                               const float* __restrict__ bias,
                                               bf16* __restrict__ obf,
                                               float* __restrict__ of32) {
    __shared__ __align__(16) bf16 xs[128 * GLD];   // [m][k]
    __shared__ __align__(16) bf16 wt[128 * GLD];   // [n][k] (W transposed)

    const int tid = threadIdx.x, lane = tid & 31, wid = tid >> 5;
    const int m0 = blockIdx.y * 128, n0 = blockIdx.x * 128;
    const int wm = wid & 1, wn = wid >> 1;   // 2x4 wave grid: 64 rows x 32 cols

    v8f acc[4][2];
#pragma unroll
    for (int mt = 0; mt < 4; ++mt)
#pragma unroll
        for (int nt = 0; nt < 2; ++nt) acc[mt][nt] = vzero();

    v4f_ xr[8], wr0[4], wr1[4];
    // prologue: stage chunk 0 into registers
#pragma unroll
    for (int t = 0; t < 8; ++t) {
        int i = tid * 4 + t * 1024, r = i >> 6, c = i & 63;
        xr[t] = *(const v4f_*)&X[(size_t)(m0 + r) * DD + c];
    }
#pragma unroll
    for (int t = 0; t < 4; ++t) {
        int u = tid + t * 256, n = (u & 31) * 4, k = (u >> 5) * 2;
        wr0[t] = *(const v4f_*)&W[(size_t)k * DD + n0 + n];
        wr1[t] = *(const v4f_*)&W[(size_t)(k + 1) * DD + n0 + n];
    }

    for (int kc = 0; kc < DD; kc += GK) {
        __syncthreads();   // all waves done reading LDS from previous stage
        // convert + store staged chunk
#pragma unroll
        for (int t = 0; t < 8; ++t) {
            int i = tid * 4 + t * 1024, r = i >> 6, c = i & 63;
            *(v4bf*)&xs[r * GLD + c] = __builtin_convertvector(xr[t], v4bf);
        }
#pragma unroll
        for (int t = 0; t < 4; ++t) {
            int u = tid + t * 256, n = (u & 31) * 4, k = (u >> 5) * 2;
#pragma unroll
            for (int e = 0; e < 4; ++e) {
                v2bf p;
                p[0] = tobf(wr0[t][e]);
                p[1] = tobf(wr1[t][e]);
                *(v2bf*)&wt[(n + e) * GLD + k] = p;
            }
        }
        __syncthreads();   // LDS tile visible

        if (kc + GK < DD) {   // prefetch next chunk; overlaps the WMMAs below
#pragma unroll
            for (int t = 0; t < 8; ++t) {
                int i = tid * 4 + t * 1024, r = i >> 6, c = i & 63;
                xr[t] = *(const v4f_*)&X[(size_t)(m0 + r) * DD + kc + GK + c];
            }
#pragma unroll
            for (int t = 0; t < 4; ++t) {
                int u = tid + t * 256, n = (u & 31) * 4, k = (u >> 5) * 2;
                wr0[t] = *(const v4f_*)&W[(size_t)(kc + GK + k) * DD + n0 + n];
                wr1[t] = *(const v4f_*)&W[(size_t)(kc + GK + k + 1) * DD + n0 + n];
            }
        }

#pragma unroll
        for (int kk = 0; kk < GK; kk += 32) {
            v16bf b0 = load_b(&wt[(wn * 32) * GLD + kk], GLD);
            v16bf b1 = load_b(&wt[(wn * 32 + 16) * GLD + kk], GLD);
#pragma unroll
            for (int mt = 0; mt < 4; ++mt) {
                v16bf af = load_a(&xs[(wm * 64 + mt * 16) * GLD + kk], GLD);
                acc[mt][0] = wmma_bf16(af, b0, acc[mt][0]);
                acc[mt][1] = wmma_bf16(af, b1, acc[mt][1]);
            }
        }
    }

    // Epilogue. C-layout: lane holds col (lane&15); VGPR j -> row j+8*(lane>=16).
    const int nl = lane & 15, r0 = (lane >> 4) << 3;
#pragma unroll
    for (int mt = 0; mt < 4; ++mt) {
#pragma unroll
        for (int nt = 0; nt < 2; ++nt) {
            int ng = n0 + wn * 32 + nt * 16 + nl;
            float bv = bias[ng];
#pragma unroll
            for (int j = 0; j < 8; ++j) {
                int m = m0 + wm * 64 + mt * 16 + r0 + j;
                float v = acc[mt][nt][j] + bv;
                if (MODE == 0 || MODE == 1) {
                    int h = ng >> 6, dh = ng & 63;
                    int b = m >> 11, s = m & 2047;
                    size_t idx = (((size_t)(h * BB + b) * SS + s) << 6) + dh;
                    obf[idx] = tobf(v);
                    if (MODE == 0) of32[idx] = v;
                } else if (MODE == 2) {
                    int h = ng >> 6, dh = ng & 63;
                    int b = m >> 11, s = m & 2047;
                    size_t idx = (((size_t)(h * BB + b) * DH + dh) * SS) + s;
                    obf[idx] = tobf(v);
                } else {  // MODE 3: residual + tanh branch
                    size_t idx = (size_t)m * DD + ng;
                    of32[idx] = X[idx] + tanhf(v);
                }
            }
        }
    }
}

// ---------------------------------------------------------------------------
// Flash attention + residual(+q) + per-head LayerNorm.
// Grid: (S/128, H*B). Block: 256 (8 waves); wave owns 16 query rows.
// K/V tiles streamed by the Tensor Data Mover with ping-pong LDS buffers:
// wave 0 issues chunk i+1's two tensor ops, then waits tensorcnt<=2 so that
// chunk i (in-order completion) is resident before the block computes on it.
// ---------------------------------------------------------------------------
#define KLD 72

__global__ __launch_bounds__(256) void attn_kernel(
    const bf16* __restrict__ q_bf, const bf16* __restrict__ k_bf,
    const bf16* __restrict__ v_t, const float* __restrict__ q_f32,
    const int* __restrict__ mask, const float* __restrict__ g_att,
    const float* __restrict__ b_att, float* __restrict__ att_ln) {
    __shared__ __align__(16) bf16 ks[2][64 * KLD];     // [key][dh]
    __shared__ __align__(16) bf16 vs[2][64 * KLD];     // [dh][key]
    __shared__ __align__(16) bf16 ps[8 * 16 * KLD];    // per-wave P tiles

    const int tid = threadIdx.x, lane = tid & 31, wid = tid >> 5;
    const int hb = blockIdx.y, h = hb >> 1, b = hb & 1;
    const int s0 = blockIdx.x * 128 + wid * 16;

    const bf16* qbase = q_bf + ((size_t)hb * SS + s0) * DH;
    v16bf qa0 = load_a(qbase, DH);         // dh 0..31
    v16bf qa1 = load_a(qbase + 32, DH);    // dh 32..63
    bf16* myps = ps + wid * 16 * KLD;
    const int* mrow = mask + b * SS;
    const bf16* kbase = k_bf + (size_t)hb * SS * DH;
    const bf16* vbase = v_t + (size_t)hb * DH * SS;

    v8f o[4];
#pragma unroll
    for (int nt = 0; nt < 4; ++nt) o[nt] = vzero();
    float mrun[8], lrun[8];
#pragma unroll
    for (int j = 0; j < 8; ++j) { mrun[j] = -3.0e38f; lrun[j] = 0.f; }

    const int nl = lane & 15, r0 = (lane >> 4) << 3;
    const int NCH = SS / 64;

    if (wid == 0) {  // kick off chunk 0 DMA
        tdm_load_2d_bf16((uint32_t)(uintptr_t)&ks[0][0], kbase,
                         64, 64, 64, SS, 64);
        tdm_load_2d_bf16((uint32_t)(uintptr_t)&vs[0][0], vbase,
                         64, 64, SS, DH, SS);
    }

    for (int it = 0; it < NCH; ++it) {
        const int kc = it * 64, cur = it & 1;
        __syncthreads();  // buf[1-cur] free: its readers (iter it-1) are done
        if (wid == 0) {
            if (it + 1 < NCH) {
                tdm_load_2d_bf16((uint32_t)(uintptr_t)&ks[1 - cur][0],
                                 kbase + (size_t)(kc + 64) * DH,
                                 64, 64, 64, SS, 64);
                tdm_load_2d_bf16((uint32_t)(uintptr_t)&vs[1 - cur][0],
                                 vbase + kc + 64,
                                 64, 64, SS, DH, SS);
                __builtin_amdgcn_s_wait_tensorcnt(2);  // chunk it resident
            } else {
                __builtin_amdgcn_s_wait_tensorcnt(0);
            }
        }
        __syncthreads();  // buf[cur] visible to all waves

        // scores 16x64: 4 N-tiles x (dh=64 -> 2 WMMA k-steps)
        v8f st[4];
#pragma unroll
        for (int nt = 0; nt < 4; ++nt) {
            st[nt] = wmma_bf16(qa0, load_b(&ks[cur][nt * 16 * KLD], KLD), vzero());
            st[nt] = wmma_bf16(qa1, load_b(&ks[cur][nt * 16 * KLD + 32], KLD), st[nt]);
        }

        // scale + additive mask
        float pv[4][8];
#pragma unroll
        for (int nt = 0; nt < 4; ++nt) {
            float ma = mrow[kc + nt * 16 + nl] ? 0.f : -1e9f;
#pragma unroll
            for (int j = 0; j < 8; ++j) pv[nt][j] = st[nt][j] * 0.125f + ma;
        }

        // online softmax (reductions across 16-lane column groups)
#pragma unroll
        for (int j = 0; j < 8; ++j) {
            float t = fmaxf(fmaxf(pv[0][j], pv[1][j]), fmaxf(pv[2][j], pv[3][j]));
            t = fmaxf(t, __shfl_xor(t, 1, 32));
            t = fmaxf(t, __shfl_xor(t, 2, 32));
            t = fmaxf(t, __shfl_xor(t, 4, 32));
            t = fmaxf(t, __shfl_xor(t, 8, 32));
            float mn = fmaxf(mrun[j], t);
            float corr = __expf(mrun[j] - mn);
            mrun[j] = mn;
            float rs = 0.f;
#pragma unroll
            for (int nt = 0; nt < 4; ++nt) {
                float p = __expf(pv[nt][j] - mn);
                pv[nt][j] = p;
                rs += p;
            }
            rs += __shfl_xor(rs, 1, 32);
            rs += __shfl_xor(rs, 2, 32);
            rs += __shfl_xor(rs, 4, 32);
            rs += __shfl_xor(rs, 8, 32);
            lrun[j] = lrun[j] * corr + rs;
#pragma unroll
            for (int nt = 0; nt < 4; ++nt) o[nt][j] *= corr;
        }

        // C-layout -> A-fragment via wave-private LDS
#pragma unroll
        for (int nt = 0; nt < 4; ++nt)
#pragma unroll
            for (int j = 0; j < 8; ++j)
                myps[(r0 + j) * KLD + nt * 16 + nl] = tobf(pv[nt][j]);

        v16bf pa0 = load_a(myps, KLD);        // keys +0..31
        v16bf pa1 = load_a(myps + 32, KLD);   // keys +32..63
#pragma unroll
        for (int nt = 0; nt < 4; ++nt) {
            o[nt] = wmma_bf16(pa0, load_b(&vs[cur][nt * 16 * KLD], KLD), o[nt]);
            o[nt] = wmma_bf16(pa1, load_b(&vs[cur][nt * 16 * KLD + 32], KLD), o[nt]);
        }
    }

    // epilogue: /l, +q residual, per-head LayerNorm over dh=64, concat heads
    float ga[4], ba[4];
#pragma unroll
    for (int nt = 0; nt < 4; ++nt) {
        ga[nt] = g_att[nt * 16 + nl];
        ba[nt] = b_att[nt * 16 + nl];
    }
#pragma unroll
    for (int j = 0; j < 8; ++j) {
        int s = s0 + r0 + j;
        float invl = 1.f / lrun[j];
        float x[4], sum = 0.f;
#pragma unroll
        for (int nt = 0; nt < 4; ++nt) {
            float qv = q_f32[(((size_t)hb * SS + s) << 6) + nt * 16 + nl];
            float vv = o[nt][j] * invl + qv;
            x[nt] = vv;
            sum += vv;
        }
        sum += __shfl_xor(sum, 1, 32);
        sum += __shfl_xor(sum, 2, 32);
        sum += __shfl_xor(sum, 4, 32);
        sum += __shfl_xor(sum, 8, 32);
        float mean = sum * (1.f / 64.f);
        float var = 0.f;
#pragma unroll
        for (int nt = 0; nt < 4; ++nt) {
            float d = x[nt] - mean;
            var += d * d;
        }
        var += __shfl_xor(var, 1, 32);
        var += __shfl_xor(var, 2, 32);
        var += __shfl_xor(var, 4, 32);
        var += __shfl_xor(var, 8, 32);
        float rstd = rsqrtf(var * (1.f / 64.f) + 1e-5f);
#pragma unroll
        for (int nt = 0; nt < 4; ++nt) {
            float y = (x[nt] - mean) * rstd * ga[nt] + ba[nt];
            att_ln[((size_t)(b * SS + s)) * DD + h * DH + nt * 16 + nl] = y;
        }
    }
}

// ---------------------------------------------------------------------------
// Final row LayerNorm over D=1024 + tanh. One block per row.
// ---------------------------------------------------------------------------
__global__ __launch_bounds__(256) void ln_tanh_kernel(
    const float* __restrict__ X, const float* __restrict__ g,
    const float* __restrict__ bb, float* __restrict__ out) {
    __shared__ float red[8];
    const int row = blockIdx.x;
    const float* x = X + (size_t)row * DD;
    const int tid = threadIdx.x, lane = tid & 31, wid = tid >> 5;

    float s = 0.f;
#pragma unroll
    for (int i = tid; i < DD; i += 256) s += x[i];
#pragma unroll
    for (int off = 1; off < 32; off <<= 1) s += __shfl_xor(s, off, 32);
    if (lane == 0) red[wid] = s;
    __syncthreads();
    float tot = 0.f;
#pragma unroll
    for (int w = 0; w < 8; ++w) tot += red[w];
    float mean = tot * (1.f / (float)DD);
    __syncthreads();

    float v = 0.f;
#pragma unroll
    for (int i = tid; i < DD; i += 256) {
        float d = x[i] - mean;
        v += d * d;
    }
#pragma unroll
    for (int off = 1; off < 32; off <<= 1) v += __shfl_xor(v, off, 32);
    if (lane == 0) red[wid] = v;
    __syncthreads();
    float vt = 0.f;
#pragma unroll
    for (int w = 0; w < 8; ++w) vt += red[w];
    float rstd = rsqrtf(vt * (1.f / (float)DD) + 1e-5f);

#pragma unroll
    for (int i = tid; i < DD; i += 256)
        out[(size_t)row * DD + i] = tanhf((x[i] - mean) * rstd * g[i] + bb[i]);
}

// ---------------------------------------------------------------------------
extern "C" void kernel_launch(void* const* d_in, const int* in_sizes, int n_in,
                              void* d_out, int out_size, void* d_ws,
                              size_t ws_size, hipStream_t stream) {
    const float* query = (const float*)d_in[0];
    const float* key   = (const float*)d_in[1];
    const float* value = (const float*)d_in[2];
    const int*   mask  = (const int*)d_in[3];
    const float* Wq = (const float*)d_in[4];
    const float* bq = (const float*)d_in[5];
    const float* Wk = (const float*)d_in[6];
    const float* bk = (const float*)d_in[7];
    const float* Wv = (const float*)d_in[8];
    const float* bv = (const float*)d_in[9];
    const float* g_att = (const float*)d_in[10];
    const float* b_att = (const float*)d_in[11];
    const float* Wres  = (const float*)d_in[12];
    const float* bres  = (const float*)d_in[13];
    const float* g_out = (const float*)d_in[14];
    const float* b_out = (const float*)d_in[15];

    char* ws = (char*)d_ws;
    const size_t MB = 1024 * 1024;
    bf16*  q_bf   = (bf16*)(ws + 0);        // 8 MB  [H,B,S,64] bf16
    bf16*  k_bf   = (bf16*)(ws + 8 * MB);   // 8 MB  [H,B,S,64] bf16
    bf16*  v_t    = (bf16*)(ws + 16 * MB);  // 8 MB  [H,B,64,S] bf16
    float* q_f32  = (float*)(ws + 24 * MB); // 16 MB [H,B,S,64] f32
    float* att_ln = (float*)(ws + 40 * MB); // 16 MB [B,S,D]    f32
    float* t_res  = (float*)(ws + 56 * MB); // 16 MB [B,S,D]    f32

    dim3 blk(256);
    dim3 gg(DD / 128, (BB * SS) / 128);     // (8, 32)

    gemm128<0><<<gg, blk, 0, stream>>>(query, Wq, bq, q_bf, q_f32);
    gemm128<1><<<gg, blk, 0, stream>>>(key, Wk, bk, k_bf, nullptr);
    gemm128<2><<<gg, blk, 0, stream>>>(value, Wv, bv, v_t, nullptr);

    attn_kernel<<<dim3(SS / 128, HH * BB), blk, 0, stream>>>(
        q_bf, k_bf, v_t, q_f32, mask, g_att, b_att, att_ln);

    gemm128<3><<<gg, blk, 0, stream>>>(att_ln, Wres, bres, nullptr, t_res);

    ln_tanh_kernel<<<dim3(BB * SS), blk, 0, stream>>>(t_res, g_out, b_out,
                                                      (float*)d_out);
}
